// Predictor_82617990906345
// MI455X (gfx1250) — matrix operand order repaired and checked by reading, not compiled, not run on hardware
//
#include <hip/hip_runtime.h>

typedef __attribute__((ext_vector_type(16))) __bf16 v16bf;
typedef __attribute__((ext_vector_type(8)))  float  v8f;
typedef __attribute__((ext_vector_type(4)))  unsigned int u32x4;
typedef unsigned short ushort_t;

union BFrag { u32x4 u[2]; v16bf bf; };

__device__ __forceinline__ ushort_t f2bf(float f) {
  unsigned u = __builtin_bit_cast(unsigned, f);
  u += 0x7FFFu + ((u >> 16) & 1u);           // round-to-nearest-even
  return (ushort_t)(u >> 16);
}

// ---------------- conversion kernels ----------------

__global__ void cvt_rows_bf16(const float* __restrict__ src,
                              ushort_t* __restrict__ dst, int n) {
  int i = blockIdx.x * blockDim.x + threadIdx.x;
  if (i < n) dst[i] = f2bf(src[i]);
}

// dst[n*K + k] = src[k*Nsrc + n] (bf16), zero-padded for n in [Nsrc, Npad)
__global__ void cvt_transpose_bf16(const float* __restrict__ src,
                                   ushort_t* __restrict__ dst,
                                   int K, int Nsrc, int Npad) {
  int i = blockIdx.x * blockDim.x + threadIdx.x;
  if (i >= K * Npad) return;
  int n = i / K, k = i - n * K;
  dst[n * K + k] = (n < Nsrc) ? f2bf(src[k * Nsrc + n]) : (ushort_t)0;
}

// ---------------- GEMM1: er = edge_ctx @ W_pe + b_pe (rows 0..9999) ----------------

__global__ __launch_bounds__(256) void gemm1_er(
    const ushort_t* __restrict__ A,    // 10000 x 512 bf16 row-major
    const ushort_t* __restrict__ Bt,   // W_pe^T: 1024 x 512 bf16
    const float*    __restrict__ bias, // b_pe[1024]
    ushort_t*       __restrict__ er)   // 10000 x 1024 bf16
{
  __shared__ __attribute__((aligned(16))) ushort_t As[16 * 520];
  const int tid  = threadIdx.x;
  const int row0 = blockIdx.x * 16;

  for (int c = tid; c < 16 * 64; c += 256) {
    int r = c >> 6, cc = (c & 63) << 3;
    *(u32x4*)&As[r * 520 + cc] = *(const u32x4*)&A[(row0 + r) * 512 + cc];
  }
  __syncthreads();

  const int wave = tid >> 5, lane = tid & 31;
  const int n = lane & 15, h = lane >> 4;

#pragma unroll 1
  for (int g = 0; g < 2; ++g) {          // 2 groups x 4 N-tiles per wave
    const ushort_t* bp[4];
#pragma unroll
    for (int t = 0; t < 4; ++t)
      bp[t] = Bt + (size_t)(wave * 128 + (g * 4 + t) * 16 + n) * 512 + 16 * h;

    BFrag bc[4], bn[4];
#pragma unroll
    for (int t = 0; t < 4; ++t) {
      bc[t].u[0] = ((const u32x4*)bp[t])[0];
      bc[t].u[1] = ((const u32x4*)bp[t])[1];
    }
    v8f acc[4] = {};
#pragma unroll
    for (int kk = 0; kk < 16; ++kk) {
      const int k0 = kk * 32;
      if (kk < 15) {
#pragma unroll
        for (int t = 0; t < 4; ++t) {
          const u32x4* gp = (const u32x4*)&bp[t][k0 + 32];
          bn[t].u[0] = gp[0]; bn[t].u[1] = gp[1];
        }
      }
      BFrag a;
      a.u[0] = *(const u32x4*)&As[n * 520 + k0 + 16 * h];
      a.u[1] = *(const u32x4*)&As[n * 520 + k0 + 16 * h + 8];
#pragma unroll
      for (int t = 0; t < 4; ++t)
        acc[t] = __builtin_amdgcn_wmma_f32_16x16x32_bf16(
            false, a.bf, false, bc[t].bf, (short)0, acc[t], false, false);
#pragma unroll
      for (int t = 0; t < 4; ++t) bc[t] = bn[t];
    }
#pragma unroll
    for (int t = 0; t < 4; ++t) {
      const int col = wave * 128 + (g * 4 + t) * 16 + n;
      const float bv = bias[col];
#pragma unroll
      for (int v = 0; v < 8; ++v)
        er[(size_t)(row0 + v + 8 * h) * 1024 + col] = f2bf(acc[t][v] + bv);
    }
  }
}

// ---------------- fused rel head (M = 48 rows per block; 150000 = 48*3125) ----------------

#define RH_STR 1032   // LDS row stride in bf16 (516 dwords = conflict-free)

__global__ __launch_bounds__(256) void relhead_fused(
    const ushort_t* __restrict__ er,     // 10000 x 1024 bf16
    const ushort_t* __restrict__ WpcT,   // 1024 x 1024 bf16 (W_pc^T)
    const ushort_t* __restrict__ WecT,   // 64 x 1024 bf16 (W_ec^T, padded)
    const ushort_t* __restrict__ WclT,   // 64 x 1024 bf16 (W_cl^T, padded)
    const float*    __restrict__ b_pc,   // [1024]
    const float*    __restrict__ b_ec,   // [51]
    const float*    __restrict__ b_cl,   // [51]
    const int*      __restrict__ pair,   // 150000 x 2
    const float*    __restrict__ uf,     // 150000 x 1024
    float*          __restrict__ out)    // 150000 x 51
{
  __shared__ __attribute__((aligned(16))) ushort_t tileS[48 * RH_STR]; // prod_rep
  __shared__ __attribute__((aligned(16))) ushort_t gateS[48 * RH_STR]; // gated
  __shared__ float redS[48 * 64];
  const int tid = threadIdx.x;
  const int R0  = blockIdx.x * 48;

  // Phase 1: async gather 48-row prod_rep tile (ASYNCcnt path); JAX clamps OOB to 9999
  for (int c = tid; c < 48 * 128; c += 256) {
    int r = c >> 7, cc = (c & 127) << 3;
    int R = R0 + r;
    int p = pair[2 * R + (cc >= 512 ? 1 : 0)];
    p = (p < 9999) ? p : 9999;
    const ushort_t* gp = &er[(size_t)p * 1024 + cc];
    unsigned ldsaddr = (unsigned)(unsigned long long)&tileS[r * RH_STR + cc];
    asm volatile("global_load_async_to_lds_b128 %0, %1, off"
                 :: "v"(ldsaddr), "v"(gp) : "memory");
  }
  asm volatile("s_wait_asynccnt 0x0" ::: "memory");
  __syncthreads();

  const int wave = tid >> 5, lane = tid & 31;
  const int n = lane & 15, h = lane >> 4;

  // Phase 2: ctx_gate = prod @ W_pc ; gate with b_pc & union_feat -> gateS (bf16)
  // Each B fragment feeds 3 M-tiles (B reuse cuts W_pc L2 traffic 3x vs M=16).
#pragma unroll 1
  for (int g = 0; g < 4; ++g) {          // 4 groups x 2 N-tiles per wave
    const ushort_t* bp[2];
#pragma unroll
    for (int t = 0; t < 2; ++t)
      bp[t] = WpcT + (size_t)(wave * 128 + (g * 2 + t) * 16 + n) * 1024 + 16 * h;

    BFrag bc[2], bn[2];
#pragma unroll
    for (int t = 0; t < 2; ++t) {
      bc[t].u[0] = ((const u32x4*)bp[t])[0];
      bc[t].u[1] = ((const u32x4*)bp[t])[1];
    }
    v8f acc[6] = {};                     // acc[t*3+mt]
#pragma unroll
    for (int kk = 0; kk < 32; ++kk) {
      const int k0 = kk * 32;
      if (kk < 31) {
#pragma unroll
        for (int t = 0; t < 2; ++t) {
          const u32x4* gp = (const u32x4*)&bp[t][k0 + 32];
          bn[t].u[0] = gp[0]; bn[t].u[1] = gp[1];
        }
      }
      BFrag a[3];
#pragma unroll
      for (int mt = 0; mt < 3; ++mt) {
        a[mt].u[0] = *(const u32x4*)&tileS[(mt * 16 + n) * RH_STR + k0 + 16 * h];
        a[mt].u[1] = *(const u32x4*)&tileS[(mt * 16 + n) * RH_STR + k0 + 16 * h + 8];
      }
#pragma unroll
      for (int t = 0; t < 2; ++t)        // 6 independent WMMA chains
#pragma unroll
        for (int mt = 0; mt < 3; ++mt)
          acc[t * 3 + mt] = __builtin_amdgcn_wmma_f32_16x16x32_bf16(
              false, a[mt].bf, false, bc[t].bf, (short)0, acc[t * 3 + mt], false, false);
#pragma unroll
      for (int t = 0; t < 2; ++t) bc[t] = bn[t];
    }
#pragma unroll
    for (int t = 0; t < 2; ++t) {
      const int col = wave * 128 + (g * 2 + t) * 16 + n;
      const float bv = b_pc[col];
#pragma unroll
      for (int mt = 0; mt < 3; ++mt) {
#pragma unroll
        for (int v = 0; v < 8; ++v) {
          const int rr = mt * 16 + v + 8 * h;
          const float gv = (acc[t * 3 + mt][v] + bv) * uf[(size_t)(R0 + rr) * 1024 + col];
          gateS[rr * RH_STR + col] = f2bf(gv);
        }
      }
    }
  }

  // Phase 3: prod @ W_cl -> redS. 12 units (3 Mtiles x 4 coltiles) over 8 waves.
  for (int u = wave; u < 12; u += 8) {
    const int mt = u >> 2, ct = u & 3;
    const int col = ct * 16 + n;
    const ushort_t* bp = WclT + (size_t)col * 1024 + 16 * h;
    v8f acc0 = {}, acc1 = {};
    BFrag bc, bn;
    bc.u[0] = ((const u32x4*)bp)[0];
    bc.u[1] = ((const u32x4*)bp)[1];
#pragma unroll
    for (int kk = 0; kk < 32; ++kk) {
      const int k0 = kk * 32;
      if (kk < 31) {
        const u32x4* gp = (const u32x4*)&bp[k0 + 32];
        bn.u[0] = gp[0]; bn.u[1] = gp[1];
      }
      BFrag a;
      a.u[0] = *(const u32x4*)&tileS[(mt * 16 + n) * RH_STR + k0 + 16 * h];
      a.u[1] = *(const u32x4*)&tileS[(mt * 16 + n) * RH_STR + k0 + 16 * h + 8];
      if (kk & 1)
        acc1 = __builtin_amdgcn_wmma_f32_16x16x32_bf16(
            false, a.bf, false, bc.bf, (short)0, acc1, false, false);
      else
        acc0 = __builtin_amdgcn_wmma_f32_16x16x32_bf16(
            false, a.bf, false, bc.bf, (short)0, acc0, false, false);
      bc = bn;
    }
    acc0 = acc0 + acc1;
#pragma unroll
    for (int v = 0; v < 8; ++v)
      redS[(mt * 16 + v + 8 * h) * 64 + col] = acc0[v];
  }
  __syncthreads();   // gateS + redS ready

  // Phase 5: gateS @ W_ec + redS + biases -> out (mask 51 cols). Same 12-unit split.
  for (int u = wave; u < 12; u += 8) {
    const int mt = u >> 2, ct = u & 3;
    const int col = ct * 16 + n;
    const ushort_t* bp = WecT + (size_t)col * 1024 + 16 * h;
    v8f acc0 = {}, acc1 = {};
    BFrag bc, bn;
    bc.u[0] = ((const u32x4*)bp)[0];
    bc.u[1] = ((const u32x4*)bp)[1];
#pragma unroll
    for (int kk = 0; kk < 32; ++kk) {
      const int k0 = kk * 32;
      if (kk < 31) {
        const u32x4* gp = (const u32x4*)&bp[k0 + 32];
        bn.u[0] = gp[0]; bn.u[1] = gp[1];
      }
      BFrag a;
      a.u[0] = *(const u32x4*)&gateS[(mt * 16 + n) * RH_STR + k0 + 16 * h];
      a.u[1] = *(const u32x4*)&gateS[(mt * 16 + n) * RH_STR + k0 + 16 * h + 8];
      if (kk & 1)
        acc1 = __builtin_amdgcn_wmma_f32_16x16x32_bf16(
            false, a.bf, false, bc.bf, (short)0, acc1, false, false);
      else
        acc0 = __builtin_amdgcn_wmma_f32_16x16x32_bf16(
            false, a.bf, false, bc.bf, (short)0, acc0, false, false);
      bc = bn;
    }
    acc0 = acc0 + acc1;
    if (col < 51) {
      const float bb = b_ec[col] + b_cl[col];
#pragma unroll
      for (int v = 0; v < 8; ++v) {
        const int rr = mt * 16 + v + 8 * h;
        out[(size_t)(R0 + rr) * 51 + col] = acc0[v] + redS[rr * 64 + col] + bb;
      }
    }
  }
}

// ---------------- launch ----------------

extern "C" void kernel_launch(void* const* d_in, const int* in_sizes, int n_in,
                              void* d_out, int out_size, void* d_ws, size_t ws_size,
                              hipStream_t stream) {
  const float* edge_ctx = (const float*)d_in[0];   // 20000x512
  const float* union_ft = (const float*)d_in[1];   // 150000x1024
  const int*   pair_idx = (const int*)d_in[2];     // 150000x2
  const float* W_pe     = (const float*)d_in[3];   // 512x1024
  const float* b_pe     = (const float*)d_in[4];   // 1024
  const float* W_pc     = (const float*)d_in[5];   // 1024x1024
  const float* b_pc     = (const float*)d_in[6];   // 1024
  const float* W_ec     = (const float*)d_in[7];   // 1024x51
  const float* b_ec     = (const float*)d_in[8];   // 51
  const float* W_cl     = (const float*)d_in[9];   // 1024x51
  const float* b_cl     = (const float*)d_in[10];  // 51
  float* out = (float*)d_out;

  char* ws = (char*)d_ws;
  const size_t OFF_EDGE = 0;                                // 10000*512*2
  const size_t OFF_WPET = OFF_EDGE + (size_t)10000*512*2;   // 1024*512*2
  const size_t OFF_WPCT = OFF_WPET + (size_t)1024*512*2;    // 1024*1024*2
  const size_t OFF_WECT = OFF_WPCT + (size_t)1024*1024*2;   // 64*1024*2
  const size_t OFF_WCLT = OFF_WECT + (size_t)64*1024*2;
  const size_t OFF_ER   = OFF_WCLT + (size_t)64*1024*2;     // 10000*1024*2

  ushort_t* edge_bf = (ushort_t*)(ws + OFF_EDGE);
  ushort_t* WpeT    = (ushort_t*)(ws + OFF_WPET);
  ushort_t* WpcT    = (ushort_t*)(ws + OFF_WPCT);
  ushort_t* WecT    = (ushort_t*)(ws + OFF_WECT);
  ushort_t* WclT    = (ushort_t*)(ws + OFF_WCLT);
  ushort_t* erB     = (ushort_t*)(ws + OFF_ER);

  {
    int n = 10000 * 512;
    cvt_rows_bf16<<<(n + 255) / 256, 256, 0, stream>>>(edge_ctx, edge_bf, n);
  }
  cvt_transpose_bf16<<<(1024 * 512  + 255) / 256, 256, 0, stream>>>(W_pe, WpeT, 512, 1024, 1024);
  cvt_transpose_bf16<<<(1024 * 1024 + 255) / 256, 256, 0, stream>>>(W_pc, WpcT, 1024, 1024, 1024);
  cvt_transpose_bf16<<<(64 * 1024   + 255) / 256, 256, 0, stream>>>(W_ec, WecT, 1024, 51, 64);
  cvt_transpose_bf16<<<(64 * 1024   + 255) / 256, 256, 0, stream>>>(W_cl, WclT, 1024, 51, 64);

  gemm1_er<<<10000 / 16, 256, 0, stream>>>(edge_bf, WpeT, b_pe, erB);

  relhead_fused<<<150000 / 48, 256, 0, stream>>>(erB, WpcT, WecT, WclT,
                                                 b_pc, b_ec, b_cl,
                                                 pair_idx, union_ft, out);
}
